// SFLayer_56075093017292
// MI455X (gfx1250) — matrix operand order repaired and checked by reading, not compiled, not run on hardware
//
#include <hip/hip_runtime.h>
#include <hip/hip_bf16.h>

// MI455X / gfx1250: wave32, WMMA 16x16x32 bf16 path.
// Sparse window-attention + submanifold-conv FFN block.
// GEMMs: 16x64 output strip per wave (A-fragment reused x4), weight panel
// staged in LDS via global_load_async_to_lds_b128 (ASYNCcnt) per block.

typedef __attribute__((ext_vector_type(16))) __bf16 v16bf;
typedef __attribute__((ext_vector_type(8)))  float  v8f;

#define NPTS 100000
#define CCH  128
#define NWIN 200
#define ROWTILES 6250          // NPTS/16 (exact)
#define GRIDY 782              // ceil(6250/8)

union AFrag { v16bf v; unsigned int u[8]; };

__device__ __forceinline__ unsigned short f32_to_bf16(float f) {
  unsigned int x = __float_as_uint(f);
  unsigned int r = (x + 0x7FFFu + ((x >> 16) & 1u)) >> 16;  // RNE
  return (unsigned short)r;
}

// ---------------------------------------------------------------------------
// Tiled GEMM: out[N x M] = A(bf16, N x K, lda) * Bt(bf16, M x K)^T
// Block: 256 threads = 8 waves, 128 rows x 64 cols per block.
// The 64-row slice of Bt (one col-strip) is staged into LDS with async
// loads; each wave computes a 16x64 strip = 4 accumulators, reusing its A
// fragment across 4 WMMAs.
// Epilogue: +bias, relu on cols < relu_limit, then one of:
//   resid != 0 : resid[row*M+col] += x   (f32 residual accumulate)
//   out32 != 0 : plain f32 store
//   else       : bf16 store to out16
// ---------------------------------------------------------------------------
template <int K>
__global__ __launch_bounds__(256) void gemm_bf16_kernel(
    const unsigned short* __restrict__ A, int lda,
    const unsigned short* __restrict__ Bt,
    const float* __restrict__ bias,
    float* __restrict__ resid,
    float* __restrict__ out32,
    unsigned short* __restrict__ out16,
    int M, int relu_limit)
{
  __shared__ __align__(16) unsigned short sB[64 * K];  // <= 32 KB

  int wave = threadIdx.x >> 5;
  int lane = threadIdx.x & 31;
  int hi   = lane >> 4;          // 0: lanes 0-15, 1: lanes 16-31
  int m    = lane & 15;
  int tile = blockIdx.y * 8 + wave;
  int nb   = blockIdx.x * 64;

  // ---- stage Bt[nb..nb+63][0..K-1] into LDS (async, per-lane 16B) ----
  {
    unsigned sbase = (unsigned)(unsigned long long)(uintptr_t)(void*)sB;
    const unsigned short* gpanel = Bt + (size_t)nb * K;
    // 8*K chunks of 16 bytes; 8*K is a multiple of 256 for K in {128,256}.
    for (int idx = threadIdx.x; idx < 8 * K; idx += 256) {
      unsigned dst = sbase + (unsigned)idx * 16u;
      unsigned long long src = (unsigned long long)(uintptr_t)(gpanel + idx * 8);
      asm volatile("global_load_async_to_lds_b128 %0, %1, off"
                   :: "v"(dst), "v"(src) : "memory");
    }
    asm volatile("s_wait_asynccnt 0x0" ::: "memory");
  }
  __syncthreads();

  if (tile >= ROWTILES) return;
  int mb = tile * 16;

  v8f acc0 = {0.f,0.f,0.f,0.f,0.f,0.f,0.f,0.f};
  v8f acc1 = acc0, acc2 = acc0, acc3 = acc0;

  const unsigned short* arow = A + (size_t)(mb + m) * lda;

  for (int kb = 0; kb < K; kb += 32) {
    AFrag a;
#pragma unroll
    for (int v = 0; v < 8; ++v) {
      int k0 = 2 * v + (hi ? 16 : 0);   // A 16x32 layout
      a.u[v] = *(const unsigned int*)(arow + kb + k0);
    }
#pragma unroll
    for (int cb = 0; cb < 4; ++cb) {
      AFrag b;
      const unsigned short* bbase = sB + (cb * 16 + m) * K + kb + (hi ? 16 : 0);
#pragma unroll
      for (int v = 0; v < 8; ++v)
        b.u[v] = *(const unsigned int*)(bbase + 2 * v);     // ds_load_b128 x2
      v8f* acc = (cb == 0) ? &acc0 : (cb == 1) ? &acc1 : (cb == 2) ? &acc2 : &acc3;
      *acc = __builtin_amdgcn_wmma_f32_16x16x32_bf16(
          false, a.v, false, b.v, (short)0, *acc, false, false);
    }
  }

#pragma unroll
  for (int cb = 0; cb < 4; ++cb) {
    const v8f& acc = (cb == 0) ? acc0 : (cb == 1) ? acc1 : (cb == 2) ? acc2 : acc3;
    int col = nb + cb * 16 + m;
    float bv = bias ? bias[col] : 0.f;
#pragma unroll
    for (int v = 0; v < 8; ++v) {
      int row = mb + v + 8 * hi;                            // C/D layout
      float x = acc[v] + bv;
      if (col < relu_limit) x = fmaxf(x, 0.f);
      size_t o = (size_t)row * M + col;
      if (resid)      resid[o] += x;
      else if (out32) out32[o] = x;
      else            out16[o] = f32_to_bf16(x);
    }
  }
}

// ---------------------------------------------------------------------------
// Submanifold gather-conv: out[:, coff:coff+32] = sum_t Fb[nbr[:,t], goff:goff+32] * W[t]
// Fb is (NPTS+1) x 128 bf16, row NPTS = zeros (sentinel). Wt: T x 32(out) x 32(in).
// One wave: 16 rows x 32 out-cols (two accumulators), one wmma pair per tap.
// ---------------------------------------------------------------------------
__global__ __launch_bounds__(256) void subm_conv_kernel(
    const unsigned short* __restrict__ Fb,
    const int* __restrict__ nbr, int T,
    const unsigned short* __restrict__ Wt,
    float* __restrict__ out, int coff, int goff)
{
  int wave = threadIdx.x >> 5;
  int lane = threadIdx.x & 31;
  int hi   = lane >> 4;
  int m    = lane & 15;
  int tile = blockIdx.x * 8 + wave;
  if (tile >= ROWTILES) return;
  int mb = tile * 16;

  v8f acc0 = {0.f,0.f,0.f,0.f,0.f,0.f,0.f,0.f};
  v8f acc1 = acc0;

  for (int t = 0; t < T; ++t) {
    int idx = nbr[(size_t)(mb + m) * T + t];                // sentinel -> zero row
    const unsigned short* arow = Fb + (size_t)idx * CCH + goff;
    const unsigned short* wb   = Wt + (size_t)t * 1024;
    AFrag a, b0, b1;
#pragma unroll
    for (int v = 0; v < 8; ++v) {
      int k0 = 2 * v + (hi ? 16 : 0);
      int kk = (hi ? 16 : 0) + 2 * v;
      a.u[v]  = *(const unsigned int*)(arow + k0);
      b0.u[v] = *(const unsigned int*)(wb + m * 32 + kk);
      b1.u[v] = *(const unsigned int*)(wb + (m + 16) * 32 + kk);
    }
    acc0 = __builtin_amdgcn_wmma_f32_16x16x32_bf16(
        false, a.v, false, b0.v, (short)0, acc0, false, false);
    acc1 = __builtin_amdgcn_wmma_f32_16x16x32_bf16(
        false, a.v, false, b1.v, (short)0, acc1, false, false);
  }
#pragma unroll
  for (int v = 0; v < 8; ++v) {
    int row = mb + v + 8 * hi;
    out[(size_t)row * CCH + coff + m]      = acc0[v];
    out[(size_t)row * CCH + coff + 16 + m] = acc1[v];
  }
}

// ------------------------------- BatchNorm ---------------------------------
__global__ __launch_bounds__(128) void bn_stats_kernel(
    const float* __restrict__ f, float* __restrict__ part)
{
  int c = threadIdx.x;
  float s = 0.f, s2 = 0.f;
  for (int n = blockIdx.x; n < NPTS; n += gridDim.x) {
    float x = f[(size_t)n * CCH + c];
    s += x; s2 += x * x;
  }
  part[(size_t)blockIdx.x * 256 + c]       = s;
  part[(size_t)blockIdx.x * 256 + 128 + c] = s2;
}

__global__ __launch_bounds__(128) void bn_finalize_kernel(
    const float* __restrict__ part, int npart,
    const float* __restrict__ g, const float* __restrict__ b,
    float* __restrict__ scale, float* __restrict__ shift)
{
  int c = threadIdx.x;
  float s = 0.f, s2 = 0.f;
  for (int p = 0; p < npart; ++p) {
    s  += part[(size_t)p * 256 + c];
    s2 += part[(size_t)p * 256 + 128 + c];
  }
  float mean = s / (float)NPTS;
  float var  = s2 / (float)NPTS - mean * mean;
  float sc   = g[c] * rsqrtf(var + 1e-3f);
  scale[c] = sc;
  shift[c] = b[c] - mean * sc;
}

// Apply BN; optional f32 out; optional bf16 out with zeroed sentinel row NPTS.
__global__ __launch_bounds__(256) void bn_apply_kernel(
    const float* __restrict__ f, const float* __restrict__ scale,
    const float* __restrict__ shift,
    float* __restrict__ o32, unsigned short* __restrict__ o16)
{
  size_t i = (size_t)blockIdx.x * 256 + threadIdx.x;
  if (i >= (size_t)(NPTS + 1) * CCH) return;
  int c = (int)(i & 127);
  int n = (int)(i >> 7);
  float x = 0.f;
  if (n < NPTS) x = f[i] * scale[c] + shift[c];
  if (o32 && n < NPTS) o32[i] = x;
  if (o16) o16[i] = f32_to_bf16(x);
}

// ------------------------------- Attention ---------------------------------
// qkv: N x 384 f32 (q/k already ReLU'd). kv: NWIN x 8 x 16 x 16, s: NWIN x 8 x 16.
__global__ __launch_bounds__(128) void attn_accum_kernel(
    const float* __restrict__ qkv, const int* __restrict__ wi,
    float* __restrict__ kv, float* __restrict__ sacc)
{
  __shared__ float vv[128];
  int n = blockIdx.x;
  int t = threadIdx.x;
  vv[t] = qkv[(size_t)n * 384 + 256 + t];
  __syncthreads();
  int w = wi[n];
  if (w < 0 || w >= NWIN) return;
  int h = t >> 4, c = t & 15;
  float k = qkv[(size_t)n * 384 + 128 + t];
  atomicAdd(&sacc[((size_t)w * 8 + h) * 16 + c], k);
  float* kvrow = kv + (((size_t)w * 8 + h) * 16 + c) * 16;
  const float* vh = vv + h * 16;
#pragma unroll
  for (int d = 0; d < 16; ++d) atomicAdd(&kvrow[d], k * vh[d]);
}

__global__ __launch_bounds__(256) void attn_apply_kernel(
    const float* __restrict__ qkv, const int* __restrict__ wi,
    const float* __restrict__ kv, const float* __restrict__ sacc,
    unsigned short* __restrict__ y16)
{
  int gid = blockIdx.x * 256 + threadIdx.x;
  int n = gid >> 3, h = gid & 7;
  if (n >= NPTS) return;
  int w = wi[n];
  if (w < 0) w = 0;
  if (w >= NWIN) w = NWIN - 1;
  const float* q   = qkv  + (size_t)n * 384 + h * 16;
  const float* sr  = sacc + ((size_t)w * 8 + h) * 16;
  const float* kvb = kv   + ((size_t)w * 8 + h) * 256;
  float y[16];
#pragma unroll
  for (int d = 0; d < 16; ++d) y[d] = 0.f;
  float z = 0.f;
  for (int c = 0; c < 16; ++c) {
    float qc = q[c];
    z += sr[c] * qc;
    const float* kvr = kvb + c * 16;
#pragma unroll
    for (int d = 0; d < 16; ++d) y[d] += qc * kvr[d];
  }
  float inv = 1.f / (z + 1e-3f);
  unsigned short* o = y16 + (size_t)n * CCH + h * 16;
#pragma unroll
  for (int d = 0; d < 16; ++d) o[d] = f32_to_bf16(y[d] * inv);
}

// ------------------------------- Misc --------------------------------------
// f = fb + (c < 96 ? conv : fb)
__global__ __launch_bounds__(256) void combine_kernel(
    const float* __restrict__ fb, const float* __restrict__ conv,
    float* __restrict__ f)
{
  size_t i = (size_t)blockIdx.x * 256 + threadIdx.x;
  if (i >= (size_t)NPTS * CCH) return;
  int c = (int)(i & 127);
  float v = fb[i];
  f[i] = v + ((c < 96) ? conv[i] : v);
}

__global__ __launch_bounds__(256) void copy_f32_kernel(
    const float* __restrict__ s, float* __restrict__ d, size_t n)
{
  size_t i = (size_t)blockIdx.x * 256 + threadIdx.x;
  if (i < n) d[i] = s[i];
}

__global__ __launch_bounds__(256) void zero_f32_kernel(float* __restrict__ p, size_t n)
{
  size_t i = (size_t)blockIdx.x * 256 + threadIdx.x;
  if (i < n) p[i] = 0.f;
}

// src: T x R x Cc f32 -> dst: T x Cc x R bf16 (transposed per tap)
__global__ __launch_bounds__(256) void convert_transpose_kernel(
    const float* __restrict__ src, unsigned short* __restrict__ dst,
    int T, int R, int Cc)
{
  size_t i = (size_t)blockIdx.x * 256 + threadIdx.x;
  size_t tot = (size_t)T * R * Cc;
  if (i >= tot) return;
  int rc = R * Cc;
  int t = (int)(i / rc);
  int rem = (int)(i % rc);
  int r = rem / Cc, c = rem % Cc;
  dst[(size_t)t * rc + (size_t)c * R + r] = f32_to_bf16(src[i]);
}

// ---------------------------------------------------------------------------
extern "C" void kernel_launch(void* const* d_in, const int* in_sizes, int n_in,
                              void* d_out, int out_size, void* d_ws, size_t ws_size,
                              hipStream_t stream) {
  (void)in_sizes; (void)n_in; (void)out_size; (void)ws_size;

  const float* feats  = (const float*)d_in[0];
  const int* win_inds = (const int*)d_in[1];
  const int* nbr_k    = (const int*)d_in[2];
  const int* nbr_h    = (const int*)d_in[3];
  const int* nbr_w    = (const int*)d_in[4];
  const float* bn_g   = (const float*)d_in[5];
  const float* bn_b   = (const float*)d_in[6];
  const float* Wqkv   = (const float*)d_in[7];
  const float* Wproj  = (const float*)d_in[8];
  const float* bproj  = (const float*)d_in[9];
  const float* bn1_g  = (const float*)d_in[10];
  const float* bn1_b  = (const float*)d_in[11];
  const float* Wk     = (const float*)d_in[12];
  const float* Wh     = (const float*)d_in[13];
  const float* Ww     = (const float*)d_in[14];
  const float* bn2_g  = (const float*)d_in[15];
  const float* bn2_b  = (const float*)d_in[16];
  const float* fc1W   = (const float*)d_in[17];
  const float* fc1b   = (const float*)d_in[18];
  const float* fc2W   = (const float*)d_in[19];
  const float* fc2b   = (const float*)d_in[20];

  // ---- workspace carve (256B aligned) ----
  char* w = (char*)d_ws;
  auto alloc = [&](size_t bytes) -> void* {
    void* p = (void*)w;
    w += (bytes + 255) & ~(size_t)255;
    return p;
  };
  float* part  = (float*)alloc(512 * 256 * sizeof(float));
  float* scale = (float*)alloc(128 * sizeof(float));
  float* shift = (float*)alloc(128 * sizeof(float));
  float* kv    = (float*)alloc((size_t)NWIN * 8 * 256 * sizeof(float));
  float* sacc  = (float*)alloc((size_t)NWIN * 8 * 16 * sizeof(float));
  float* bufA  = (float*)alloc((size_t)NPTS * 384 * sizeof(float));     // qkv / fb(f32) / fc1out(bf16)
  unsigned short* bufB = (unsigned short*)alloc((size_t)(NPTS + 1) * CCH * 2); // bf16 activations (+sentinel row)
  float* bufC  = (float*)alloc((size_t)NPTS * CCH * sizeof(float));     // conv output
  unsigned short* WqkvT  = (unsigned short*)alloc((size_t)2 * 384 * 128 * 2);
  unsigned short* WprojT = (unsigned short*)alloc((size_t)2 * 128 * 128 * 2);
  unsigned short* fc1T   = (unsigned short*)alloc((size_t)2 * 256 * 128 * 2);
  unsigned short* fc2T   = (unsigned short*)alloc((size_t)2 * 128 * 256 * 2);
  unsigned short* WkT    = (unsigned short*)alloc((size_t)2 * 27 * 1024 * 2);
  unsigned short* WhT    = (unsigned short*)alloc((size_t)2 * 13 * 1024 * 2);
  unsigned short* WwT    = (unsigned short*)alloc((size_t)2 * 13 * 1024 * 2);

  float* f = (float*)d_out;  // running feature buffer lives in d_out

  auto gblk = [](size_t n) { return (unsigned)((n + 255) / 256); };

  // f = feats
  copy_f32_kernel<<<gblk((size_t)NPTS * CCH), 256, 0, stream>>>(feats, f, (size_t)NPTS * CCH);

  // weights -> transposed bf16
  auto ct = [&](const float* src, unsigned short* dst, int T, int R, int Cc) {
    size_t tot = (size_t)T * R * Cc;
    convert_transpose_kernel<<<gblk(tot), 256, 0, stream>>>(src, dst, T, R, Cc);
  };
  ct(Wqkv,  WqkvT,  2, 128, 384);
  ct(Wproj, WprojT, 2, 128, 128);
  ct(fc1W,  fc1T,   2, 128, 256);
  ct(fc2W,  fc2T,   2, 256, 128);
  ct(Wk, WkT, 2 * 27, 32, 32);
  ct(Wh, WhT, 2 * 13, 32, 32);
  ct(Ww, WwT, 2 * 13, 32, 32);

  unsigned bnapply_blocks = gblk((size_t)(NPTS + 1) * CCH);

  for (int d = 0; d < 2; ++d) {
    const int* wi = win_inds + (size_t)d * NPTS;

    // ---- BN0 + attention ----
    bn_stats_kernel<<<512, 128, 0, stream>>>(f, part);
    bn_finalize_kernel<<<1, 128, 0, stream>>>(part, 512, bn_g + d * 128, bn_b + d * 128, scale, shift);
    bn_apply_kernel<<<bnapply_blocks, 256, 0, stream>>>(f, scale, shift, nullptr, bufB);

    // qkv = bn(f) @ Wqkv ; relu on q,k (cols < 256)
    gemm_bf16_kernel<128><<<dim3(384 / 64, GRIDY), 256, 0, stream>>>(
        bufB, 128, WqkvT + (size_t)d * 384 * 128, nullptr, nullptr, bufA, nullptr,
        384, 256);

    zero_f32_kernel<<<gblk((size_t)NWIN * 8 * 256), 256, 0, stream>>>(kv, (size_t)NWIN * 8 * 256);
    zero_f32_kernel<<<gblk((size_t)NWIN * 8 * 16), 256, 0, stream>>>(sacc, (size_t)NWIN * 8 * 16);
    attn_accum_kernel<<<NPTS, 128, 0, stream>>>(bufA, wi, kv, sacc);
    attn_apply_kernel<<<(NPTS * 8) / 256, 256, 0, stream>>>(bufA, wi, kv, sacc, bufB);

    // f += attn @ Wproj + bproj
    gemm_bf16_kernel<128><<<dim3(128 / 64, GRIDY), 256, 0, stream>>>(
        bufB, 128, WprojT + (size_t)d * 128 * 128, bproj + d * 128, f, nullptr, nullptr,
        128, 0);

    // ---- BN1 + submanifold convs ----
    bn_stats_kernel<<<512, 128, 0, stream>>>(f, part);
    bn_finalize_kernel<<<1, 128, 0, stream>>>(part, 512, bn1_g + d * 128, bn1_b + d * 128, scale, shift);
    bn_apply_kernel<<<bnapply_blocks, 256, 0, stream>>>(f, scale, shift, bufA, bufB);

    subm_conv_kernel<<<GRIDY, 256, 0, stream>>>(bufB, nbr_k, 27, WkT + (size_t)d * 27 * 1024, bufC, 0, 0);
    subm_conv_kernel<<<GRIDY, 256, 0, stream>>>(bufB, nbr_h, 13, WhT + (size_t)d * 13 * 1024, bufC, 32, 32);
    subm_conv_kernel<<<GRIDY, 256, 0, stream>>>(bufB, nbr_w, 13, WwT + (size_t)d * 13 * 1024, bufC, 64, 64);
    combine_kernel<<<gblk((size_t)NPTS * CCH), 256, 0, stream>>>(bufA, bufC, f);

    // ---- BN2 + FFN ----
    bn_stats_kernel<<<512, 128, 0, stream>>>(f, part);
    bn_finalize_kernel<<<1, 128, 0, stream>>>(part, 512, bn2_g + d * 128, bn2_b + d * 128, scale, shift);
    bn_apply_kernel<<<bnapply_blocks, 256, 0, stream>>>(f, scale, shift, nullptr, bufB);

    unsigned short* fc1out = (unsigned short*)bufA;  // N x 256 bf16
    gemm_bf16_kernel<128><<<dim3(256 / 64, GRIDY), 256, 0, stream>>>(
        bufB, 128, fc1T + (size_t)d * 256 * 128, fc1b + d * 256, nullptr, nullptr, fc1out,
        256, 256);
    gemm_bf16_kernel<256><<<dim3(128 / 64, GRIDY), 256, 0, stream>>>(
        fc1out, 256, fc2T + (size_t)d * 128 * 256, fc2b + d * 128, f, nullptr, nullptr,
        128, 0);
  }
}
